// feature_aug_25142738550823
// MI455X (gfx1250) — matrix operand order repaired and checked by reading, not compiled, not run on hardware
//
#include <hip/hip_runtime.h>

// ---------------------------------------------------------------------------
// Problem constants (from the reference):
//   N=32, K=128, C=2048, H=W=24 -> HW=S=576, D=C/2=1024
// All GEMMs run on v_wmma_f32_16x16x32_bf16 (fp32 accumulate), with operands
// pre-packed into contiguous-k WMMA lane layouts (same k order for A and B) so the main loop is pure
// global_load_b128 + v_wmma, ping-pong software-pipelined (no register copies).
// ---------------------------------------------------------------------------

typedef __attribute__((ext_vector_type(16))) __bf16 v16bf;
typedef __attribute__((ext_vector_type(8)))  float  v8f;

__device__ __forceinline__ unsigned short f2bf(float f) {
    union { float f; unsigned u; } v; v.f = f;
    unsigned r = v.u + 0x7FFFu + ((v.u >> 16) & 1u);   // round-to-nearest-even
    return (unsigned short)(r >> 16);
}

// ---------------------------------------------------------------------------
// Packing kernels: produce fragment-major operands in a contiguous-k WMMA
// lane layout (same k order for A and B) so the GEMM main loop does pure b128 loads.
//
// A tile (16x32 bf16), lane l, slot i:  m = l%16,  k = (l/16)*16 + i
// B tile (32x16 bf16), lane l, slot i:  n = l%16,  k = (l/16)*16 + i
// Packed storage: tile-major, 32 lanes x 16 bf16 contiguous per tile.
// ---------------------------------------------------------------------------

__global__ void pack_a_f32(const float* __restrict__ src, unsigned short* __restrict__ dst,
                           int MT, int KT, int KT_ld, int kt0, int ld, int transposed,
                           long srcBatch, long dstBatch) {
    int t = blockIdx.x * blockDim.x + threadIdx.x;
    int tile = t >> 5, lane = t & 31;
    if (tile >= MT * KT) return;
    int mt = tile / KT, kt = tile % KT;
    const float* s = src + (size_t)blockIdx.z * srcBatch;
    unsigned short* d = dst + (size_t)blockIdx.z * dstBatch
                      + ((size_t)(mt * KT_ld + kt0 + kt) * 32 + lane) * 16;
    int m  = mt * 16 + (lane & 15);
    int kb = kt * 32 + ((lane >> 4) * 16);
#pragma unroll
    for (int i = 0; i < 16; ++i) {
        int k = kb + i;
        float v = transposed ? s[(size_t)k * ld + m] : s[(size_t)m * ld + k];
        d[i] = f2bf(v);
    }
}

__global__ void pack_a_bf16(const unsigned short* __restrict__ src, unsigned short* __restrict__ dst,
                            int MT, int KT, int KT_ld, int kt0, int ld,
                            long srcBatch, long dstBatch) {
    int t = blockIdx.x * blockDim.x + threadIdx.x;
    int tile = t >> 5, lane = t & 31;
    if (tile >= MT * KT) return;
    int mt = tile / KT, kt = tile % KT;
    const unsigned short* s = src + (size_t)blockIdx.z * srcBatch;
    unsigned short* d = dst + (size_t)blockIdx.z * dstBatch
                      + ((size_t)(mt * KT_ld + kt0 + kt) * 32 + lane) * 16;
    int m  = mt * 16 + (lane & 15);
    int kb = kt * 32 + ((lane >> 4) * 16);
#pragma unroll
    for (int i = 0; i < 16; ++i) {
        int k = kb + i;
        d[i] = s[(size_t)m * ld + k];
    }
}

__global__ void pack_b_f32(const float* __restrict__ src, unsigned short* __restrict__ dst,
                           int KT, int NT, int ld, long srcBatch, long dstBatch) {
    int t = blockIdx.x * blockDim.x + threadIdx.x;
    int tile = t >> 5, lane = t & 31;
    if (tile >= KT * NT) return;
    int kt = tile / NT, nt = tile % NT;
    const float* s = src + (size_t)blockIdx.z * srcBatch;
    unsigned short* d = dst + (size_t)blockIdx.z * dstBatch
                      + ((size_t)tile * 32 + lane) * 16;
    int n  = nt * 16 + (lane & 15);
    int kb = kt * 32 + ((lane >> 4) * 16);
#pragma unroll
    for (int i = 0; i < 16; ++i)
        d[i] = f2bf(s[(size_t)(kb + i) * ld + n]);
}

__global__ void pack_b_bf16(const unsigned short* __restrict__ src, unsigned short* __restrict__ dst,
                            int KT, int NT, int ld, long srcBatch, long dstBatch) {
    int t = blockIdx.x * blockDim.x + threadIdx.x;
    int tile = t >> 5, lane = t & 31;
    if (tile >= KT * NT) return;
    int kt = tile / NT, nt = tile % NT;
    const unsigned short* s = src + (size_t)blockIdx.z * srcBatch;
    unsigned short* d = dst + (size_t)blockIdx.z * dstBatch
                      + ((size_t)tile * 32 + lane) * 16;
    int n  = nt * 16 + (lane & 15);
    int kb = kt * 32 + ((lane >> 4) * 16);
#pragma unroll
    for (int i = 0; i < 16; ++i)
        d[i] = s[(size_t)(kb + i) * ld + n];
}

// B^T pack: B[k,n] = src[n*ld + k]   (used for logits = fproj @ pproj^T)
__global__ void pack_bt_bf16(const unsigned short* __restrict__ src, unsigned short* __restrict__ dst,
                             int KT, int NT, int ld, long srcBatch, long dstBatch) {
    int t = blockIdx.x * blockDim.x + threadIdx.x;
    int tile = t >> 5, lane = t & 31;
    if (tile >= KT * NT) return;
    int kt = tile / NT, nt = tile % NT;
    const unsigned short* s = src + (size_t)blockIdx.z * srcBatch;
    unsigned short* d = dst + (size_t)blockIdx.z * dstBatch
                      + ((size_t)tile * 32 + lane) * 16;
    int n  = nt * 16 + (lane & 15);
    int kb = kt * 32 + ((lane >> 4) * 16);
#pragma unroll
    for (int i = 0; i < 16; ++i)
        d[i] = s[(size_t)n * ld + (kb + i)];
}

// ---------------------------------------------------------------------------
// Generic bf16 WMMA GEMM.  Block = 128 threads = 4 waves; each wave owns a
// 16(M) x 64(N) output strip => 4 accumulators, 4 v_wmma per K-step of 32.
// Ping-pong pipelined: two disjoint fragment sets alternate roles across an
// unroll-by-2 loop so the register allocator emits zero buffer-copy movs and
// the loads for step k+1 stay in flight under the WMMAs of step k.
// Requires KT even and >= 2 (all call sites: 64, 32, 4).
// mode 0: out_bf16 row-major (+optional ReLU)
// mode 1: out_f32  row-major
// mode 2: fused final epilogue: relu(x + acc(+bias)) -> NCHW d_out
// ---------------------------------------------------------------------------
__global__ __launch_bounds__(128)
void gemm_bf16_wmma(const unsigned short* __restrict__ Ap,
                    const unsigned short* __restrict__ Bp,
                    const float* __restrict__ bias,
                    unsigned short* __restrict__ out_bf16,
                    float* __restrict__ out_f32,
                    const float* __restrict__ resid_x,
                    float* __restrict__ resid_out,
                    int MT, int NT, int KT, int KT_ld, int kt0,
                    int Nld, int HW, int relu, int mode,
                    long aBatch, long bBatch, long cBatch) {
    const int lane = threadIdx.x & 31;
    const int wave = threadIdx.x >> 5;
    const int mt = blockIdx.y * 4 + wave;
    if (mt >= MT) return;                     // wave-uniform: EXEC stays all-ones
    const int nt0 = blockIdx.x * 4;
    const int b = blockIdx.z;
    const unsigned short* A = Ap + (size_t)b * aBatch;
    const unsigned short* B = Bp + (size_t)b * bBatch;
    const int nl = lane & 15;

    v8f acc[4];
#pragma unroll
    for (int j = 0; j < 4; ++j) {
        float bv = bias ? bias[(nt0 + j) * 16 + nl] : 0.0f;
#pragma unroll
        for (int r = 0; r < 8; ++r) acc[j][r] = bv;
    }

    // Fragment pointers: A tiles contiguous along kt (512 u16 per tile);
    // B kt-stride = NT tiles.
    const unsigned short* aptr = A + ((size_t)(mt * KT_ld + kt0) * 32 + lane) * 16;
    const unsigned short* bptr = B + ((size_t)nt0 * 32 + lane) * 16;
    const size_t bStep = (size_t)NT * 512;

    // Ping-pong fragment sets (never assigned to each other).
    v16bf a0, a1, b0[4], b1[4];

    // Prologue: stage 0 holds kt = 0.
    a0 = *(const v16bf*)aptr;
#pragma unroll
    for (int j = 0; j < 4; ++j)
        b0[j] = *(const v16bf*)(bptr + (size_t)j * 512);

    // Main loop: 2 K-steps per iteration, KT even.
    for (int kt = 0; kt < KT - 2; kt += 2) {
        // load kt+1 into stage 1
        a1 = *(const v16bf*)(aptr + 512);
#pragma unroll
        for (int j = 0; j < 4; ++j)
            b1[j] = *(const v16bf*)(bptr + bStep + (size_t)j * 512);
        // compute kt (stage 0)
#pragma unroll
        for (int j = 0; j < 4; ++j)
            acc[j] = __builtin_amdgcn_wmma_f32_16x16x32_bf16(
                false, a0, false, b0[j], (short)0, acc[j], false, false);
        // load kt+2 into stage 0
        a0 = *(const v16bf*)(aptr + 1024);
#pragma unroll
        for (int j = 0; j < 4; ++j)
            b0[j] = *(const v16bf*)(bptr + 2 * bStep + (size_t)j * 512);
        // compute kt+1 (stage 1)
#pragma unroll
        for (int j = 0; j < 4; ++j)
            acc[j] = __builtin_amdgcn_wmma_f32_16x16x32_bf16(
                false, a1, false, b1[j], (short)0, acc[j], false, false);
        aptr += 1024;
        bptr += 2 * bStep;
    }
    // Epilogue: steps KT-2 (stage 0, already loaded) and KT-1.
    a1 = *(const v16bf*)(aptr + 512);
#pragma unroll
    for (int j = 0; j < 4; ++j)
        b1[j] = *(const v16bf*)(bptr + bStep + (size_t)j * 512);
#pragma unroll
    for (int j = 0; j < 4; ++j)
        acc[j] = __builtin_amdgcn_wmma_f32_16x16x32_bf16(
            false, a0, false, b0[j], (short)0, acc[j], false, false);
#pragma unroll
    for (int j = 0; j < 4; ++j)
        acc[j] = __builtin_amdgcn_wmma_f32_16x16x32_bf16(
            false, a1, false, b1[j], (short)0, acc[j], false, false);

    // C/D layout: vgpr r, lanes 0-15 -> M=r, N=lane; lanes 16-31 -> M=8+r
    const int mg = (lane >> 4) * 8;
#pragma unroll
    for (int j = 0; j < 4; ++j) {
        int n = (nt0 + j) * 16 + nl;
#pragma unroll
        for (int r = 0; r < 8; ++r) {
            int m = mt * 16 + mg + r;
            float v = acc[j][r];
            if (mode == 0) {
                if (relu) v = fmaxf(v, 0.0f);
                out_bf16[(size_t)b * cBatch + (size_t)m * Nld + n] = f2bf(v);
            } else if (mode == 1) {
                out_f32[(size_t)b * cBatch + (size_t)m * Nld + n] = v;
            } else {
                // final: out[n,c,s] = relu(x[n,c,s] + agg@W4 + b4); c=n, s=m
                size_t idx = (size_t)b * cBatch + (size_t)n * HW + m;
                v += resid_x[idx];
                resid_out[idx] = fmaxf(v, 0.0f);
            }
        }
    }
}

// ---------------------------------------------------------------------------
// Masked softmax over K=128 prototypes: one wave per (n, s) row.
// ---------------------------------------------------------------------------
__global__ __launch_bounds__(128)
void softmax_att(const float* __restrict__ logits, const int* __restrict__ labels,
                 unsigned short* __restrict__ att, int S, int Kdim) {
    int lane = threadIdx.x & 31;
    int row  = blockIdx.x * (blockDim.x >> 5) + (threadIdx.x >> 5);
    int n = row / S, s = row % S;
    const float* lg = logits + ((size_t)n * S + s) * Kdim;
    const int* lab = labels + (size_t)n * Kdim;
    float v[4];
#pragma unroll
    for (int j = 0; j < 4; ++j) {
        int k = lane + j * 32;
        v[j] = (lab[k] > 0) ? lg[k] : -3.0e38f;
    }
    float mx = fmaxf(fmaxf(v[0], v[1]), fmaxf(v[2], v[3]));
#pragma unroll
    for (int off = 16; off >= 1; off >>= 1) mx = fmaxf(mx, __shfl_xor(mx, off, 32));
    float e[4], sum = 0.0f;
#pragma unroll
    for (int j = 0; j < 4; ++j) { e[j] = __expf(v[j] - mx); sum += e[j]; }
#pragma unroll
    for (int off = 16; off >= 1; off >>= 1) sum += __shfl_xor(sum, off, 32);
    float inv = 1.0f / sum;
    unsigned short* o = att + ((size_t)n * S + s) * Kdim;
#pragma unroll
    for (int j = 0; j < 4; ++j) o[lane + j * 32] = f2bf(e[j] * inv);
}

// ---------------------------------------------------------------------------
extern "C" void kernel_launch(void* const* d_in, const int* in_sizes, int n_in,
                              void* d_out, int out_size, void* d_ws, size_t ws_size,
                              hipStream_t stream) {
    (void)in_sizes; (void)n_in; (void)out_size; (void)ws_size;

    const int NB = 32, KP = 128, C = 2048, HW = 576, D = 1024, S = 576;

    const float* x          = (const float*)d_in[0];
    const float* prototypes = (const float*)d_in[1];
    const int*   labels     = (const int*)d_in[2];
    const float* Wp = (const float*)d_in[3];  const float* bp  = (const float*)d_in[4];
    const float* Wf = (const float*)d_in[5];  const float* bfv = (const float*)d_in[6];
    const float* W3 = (const float*)d_in[7];  const float* b3  = (const float*)d_in[8];
    const float* W4 = (const float*)d_in[9];  const float* b4  = (const float*)d_in[10];
    float* out = (float*)d_out;

    // --- workspace sub-allocation (256B aligned) ---
    char* wsp = (char*)d_ws;
    size_t off = 0;
    auto alloc = [&](size_t bytes) -> char* {
        char* p = wsp + off;
        off = (off + bytes + 255) & ~(size_t)255;
        return p;
    };
    unsigned short* featA   = (unsigned short*)alloc((size_t)NB * S * C * 2);   // A pack, M=576 K=2048
    unsigned short* protoA  = (unsigned short*)alloc((size_t)NB * KP * C * 2);  // A pack, M=128 K=2048
    unsigned short* WpB     = (unsigned short*)alloc((size_t)C * D * 2);
    unsigned short* WfB     = (unsigned short*)alloc((size_t)C * D * 2);
    unsigned short* W3B     = (unsigned short*)alloc((size_t)C * D * 2);
    unsigned short* W4B     = (unsigned short*)alloc((size_t)D * C * 2);
    unsigned short* pprojRM = (unsigned short*)alloc((size_t)NB * KP * D * 2);  // bf16 row-major
    unsigned short* fprojRM = (unsigned short*)alloc((size_t)NB * S * D * 2);
    unsigned short* protoB  = (unsigned short*)alloc((size_t)NB * KP * D * 2);  // B pack (proto_aug)
    unsigned short* protoBT = (unsigned short*)alloc((size_t)NB * KP * D * 2);  // B^T pack (logits)
    unsigned short* catA    = (unsigned short*)alloc((size_t)NB * S * C * 2);   // [proto_aug | feat_proj]
    float*          logitsW = (float*)alloc((size_t)NB * S * KP * 4);
    unsigned short* attRM   = (unsigned short*)alloc((size_t)NB * S * KP * 2);
    unsigned short* attA    = (unsigned short*)alloc((size_t)NB * S * KP * 2);
    unsigned short* paugRM  = (unsigned short*)alloc((size_t)NB * S * D * 2);
    unsigned short* aggRM   = (unsigned short*)alloc((size_t)NB * S * D * 2);
    unsigned short* aggA    = (unsigned short*)alloc((size_t)NB * S * D * 2);

    // --- stage 0: pack fp32 inputs to bf16 fragment-major operands ---
    // features: x is [n, c, hw] -> A rows are s (transposed gather)
    pack_a_f32<<<dim3(288, 1, NB), 256, 0, stream>>>(
        x, featA, 36, 64, 64, 0, /*ld=*/HW, /*transposed=*/1,
        (long)C * HW, (long)S * C);
    pack_a_f32<<<dim3(64, 1, NB), 256, 0, stream>>>(
        prototypes, protoA, 8, 64, 64, 0, /*ld=*/C, 0,
        (long)KP * C, (long)KP * C);
    pack_b_f32<<<dim3(512, 1, 1), 256, 0, stream>>>(Wp, WpB, 64, 64, D, 0, 0);
    pack_b_f32<<<dim3(512, 1, 1), 256, 0, stream>>>(Wf, WfB, 64, 64, D, 0, 0);
    pack_b_f32<<<dim3(512, 1, 1), 256, 0, stream>>>(W3, W3B, 64, 64, D, 0, 0);
    pack_b_f32<<<dim3(512, 1, 1), 256, 0, stream>>>(W4, W4B, 32, 128, C, 0, 0);

    // --- GEMM1: proto_proj = prototypes @ Wp + bp   [128 x 1024] ---
    gemm_bf16_wmma<<<dim3(4, 2, NB), 128, 0, stream>>>(
        protoA, WpB, bp, pprojRM, nullptr, nullptr, nullptr,
        8, 16, 64, 64, 0, D, 0, 0, 0, (long)KP * C, 0, (long)KP * D);

    // --- GEMM2: feat_proj = features @ Wf + bf   [576 x 1024] ---
    gemm_bf16_wmma<<<dim3(4, 9, NB), 128, 0, stream>>>(
        featA, WfB, bfv, fprojRM, nullptr, nullptr, nullptr,
        36, 16, 64, 64, 0, D, 0, 0, 0, (long)S * C, 0, (long)S * D);

    // repack proto_proj as B (for proto_aug) and B^T (for logits)
    pack_b_bf16<<<dim3(256, 1, NB), 256, 0, stream>>>(
        pprojRM, protoB, 32, 64, D, (long)KP * D, (long)KP * D);
    pack_bt_bf16<<<dim3(32, 1, NB), 256, 0, stream>>>(
        pprojRM, protoBT, 32, 8, D, (long)KP * D, (long)KP * D);
    // feat_proj -> upper K-half of catA (kt offset 32)
    pack_a_bf16<<<dim3(144, 1, NB), 256, 0, stream>>>(
        fprojRM, catA, 36, 32, 64, 32, D, (long)S * D, (long)S * C);

    // --- GEMM3: logits = feat_proj @ proto_proj^T   [576 x 128] fp32 out ---
    gemm_bf16_wmma<<<dim3(2, 9, NB), 128, 0, stream>>>(
        catA, protoBT, nullptr, nullptr, logitsW, nullptr, nullptr,
        36, 8, 32, 64, 32, KP, 0, 0, 1, (long)S * C, (long)KP * D, (long)S * KP);

    // --- masked softmax over K ---
    softmax_att<<<dim3((NB * S) / 4, 1, 1), 128, 0, stream>>>(logitsW, labels, attRM, S, KP);
    pack_a_bf16<<<dim3(18, 1, NB), 256, 0, stream>>>(
        attRM, attA, 36, 4, 4, 0, KP, (long)S * KP, (long)S * KP);

    // --- GEMM4: proto_aug = att @ proto_proj   [576 x 1024] ---
    gemm_bf16_wmma<<<dim3(4, 9, NB), 128, 0, stream>>>(
        attA, protoB, nullptr, paugRM, nullptr, nullptr, nullptr,
        36, 16, 4, 4, 0, D, 0, 0, 0, (long)S * KP, (long)KP * D, (long)S * D);
    // proto_aug -> lower K-half of catA
    pack_a_bf16<<<dim3(144, 1, NB), 256, 0, stream>>>(
        paugRM, catA, 36, 32, 64, 0, D, (long)S * D, (long)S * C);

    // --- GEMM5: agg = relu(cat @ W3 + b3)   [576 x 1024] ---
    gemm_bf16_wmma<<<dim3(4, 9, NB), 128, 0, stream>>>(
        catA, W3B, b3, aggRM, nullptr, nullptr, nullptr,
        36, 16, 64, 64, 0, D, 0, 1, 0, (long)S * C, 0, (long)S * D);
    pack_a_bf16<<<dim3(144, 1, NB), 256, 0, stream>>>(
        aggRM, aggA, 36, 32, 32, 0, D, (long)S * D, (long)S * D);

    // --- GEMM6: out = relu(x + agg @ W4 + b4), fused NCHW epilogue ---
    gemm_bf16_wmma<<<dim3(8, 9, NB), 128, 0, stream>>>(
        aggA, W4B, b4, nullptr, nullptr, x, out,
        36, 32, 32, 32, 0, C, HW, 1, 2, (long)S * D, 0, (long)C * HW);
}
